// Qwen3VLVisionAttention_27668179321435
// MI455X (gfx1250) — matrix operand order repaired and checked by reading, not compiled, not run on hardware
//
#include <hip/hip_runtime.h>

// ---------------------------------------------------------------------------
// Qwen3-VL vision attention block on MI455X (gfx1250, CDNA5, wave32).
// Pipeline:
//   f32->f16 converts (weights stored pre-transposed W^T[N][K])
//   -> QKV GEMM: WMMA f16/f32-accum, double-buffered LDS tiles filled with
//      GLOBAL_LOAD_ASYNC_TO_LDS_B128 (ASYNCcnt pipelined), fused bias +
//      scatter to padded per-head Q/K and transposed V
//   -> RoPE (f32 math on f16 Q/K)
//   -> per-(seg,head,64-query) attention with the full 64x1024 fp32 score
//      strip resident in CDNA5's 320KB LDS (WMMA QK^T, wave32 ds_swizzle
//      softmax reductions, WMMA PV)
//   -> output projection GEMM (same async/WMMA path) -> fp32 d_out.
// ---------------------------------------------------------------------------

#define DIMV   1280
#define HEADSV 16
#define HDV    80
#define HDP    96            // head dim padded to 3*32 for WMMA K-steps
#define NSEGV  8
#define LSEGV  1024
#define SEQV   8192
#define NQKV   (3 * DIMV)

typedef __attribute__((ext_vector_type(16))) _Float16 v16h;
typedef __attribute__((ext_vector_type(8)))  float    v8f;

union Frag16 { v16h h; uint4 q[2]; };

// ---- CDNA5 async global->LDS copy (VGLOBAL async, GVS mode, ASYNCcnt) -----
// global_load_async_to_lds_b128: VDST = LDS byte address, VADDR = 32-bit
// per-lane offset, SADDR = uniform 64-bit base. Generic pointers to __shared__
// carry the LDS byte offset in their low 32 bits on AMDGPU.
__device__ __forceinline__ void async_ld16(void* lds, const void* gbase,
                                           int voff) {
  unsigned lds_off = (unsigned)(uintptr_t)lds;
  asm volatile("global_load_async_to_lds_b128 %0, %1, %2"
               :
               : "v"(lds_off), "v"(voff), "s"(gbase)
               : "memory");
}
__device__ __forceinline__ void wait_async0() {
  asm volatile("s_wait_asynccnt 0x0" ::: "memory");
}
__device__ __forceinline__ void wait_async4() {
  asm volatile("s_wait_asynccnt 0x4" ::: "memory");
}

// wave32 xor-reductions via ds_swizzle (group-of-32: imm = (xor<<10)|0x1F)
__device__ __forceinline__ float wave_reduce_max(float v) {
  v = fmaxf(v, __int_as_float(__builtin_amdgcn_ds_swizzle(__float_as_int(v), 0x401F)));
  v = fmaxf(v, __int_as_float(__builtin_amdgcn_ds_swizzle(__float_as_int(v), 0x201F)));
  v = fmaxf(v, __int_as_float(__builtin_amdgcn_ds_swizzle(__float_as_int(v), 0x101F)));
  v = fmaxf(v, __int_as_float(__builtin_amdgcn_ds_swizzle(__float_as_int(v), 0x081F)));
  v = fmaxf(v, __int_as_float(__builtin_amdgcn_ds_swizzle(__float_as_int(v), 0x041F)));
  return v;
}
__device__ __forceinline__ float wave_reduce_sum(float v) {
  v += __int_as_float(__builtin_amdgcn_ds_swizzle(__float_as_int(v), 0x401F));
  v += __int_as_float(__builtin_amdgcn_ds_swizzle(__float_as_int(v), 0x201F));
  v += __int_as_float(__builtin_amdgcn_ds_swizzle(__float_as_int(v), 0x101F));
  v += __int_as_float(__builtin_amdgcn_ds_swizzle(__float_as_int(v), 0x081F));
  v += __int_as_float(__builtin_amdgcn_ds_swizzle(__float_as_int(v), 0x041F));
  return v;
}

// ---------------------------- elementwise converts --------------------------
__global__ void k_f32_to_f16(const float* __restrict__ in,
                             _Float16* __restrict__ out, int n) {
  int i = blockIdx.x * 256 + threadIdx.x;
  if (i < n) out[i] = (_Float16)in[i];
}

// weights: f32 [K][N] -> f16 transposed [N][K] (coalesced reads)
__global__ void k_f32_to_f16_T(const float* __restrict__ in,
                               _Float16* __restrict__ out, int Kd, int Nd) {
  int i = blockIdx.x * 256 + threadIdx.x;
  if (i < Kd * Nd) {
    int k = i / Nd, n = i - k * Nd;
    out[(size_t)n * Kd + k] = (_Float16)in[i];
  }
}

// ------------------- zero pad columns [80,96) of Q and K --------------------
__global__ void k_pad_qk(_Float16* __restrict__ Q, _Float16* __restrict__ K) {
  int i = blockIdx.x * 256 + threadIdx.x;   // 2 * 131072 * 16 threads
  int col = 80 + (i & 15);
  int row = (i >> 4) & (131072 - 1);        // 8*16*1024 = 2^17 rows
  int buf = i >> 21;
  _Float16* p = buf ? K : Q;
  p[(size_t)row * HDP + col] = (_Float16)0.0f;
}

// ------------------------------- RoPE (f32 math) ----------------------------
__global__ void k_rope(_Float16* __restrict__ Q, _Float16* __restrict__ K,
                       const float* __restrict__ cosT,
                       const float* __restrict__ sinT) {
  int i = blockIdx.x * 256 + threadIdx.x;   // 2*8*16*1024*40 threads
  int d = i % 40; int rest = i / 40;
  int row  = rest & 1023; rest >>= 10;
  int head = rest & 15;   rest >>= 4;
  int seg  = rest & 7;    int buf = rest >> 3;
  _Float16* p = buf ? K : Q;
  int pos = seg * LSEGV + row;
  size_t base = ((size_t)(seg * HEADSV + head) * LSEGV + row) * HDP;
  float x1 = (float)p[base + d];
  float x2 = (float)p[base + d + 40];
  const float* cr = cosT + (size_t)pos * HDV;
  const float* sr = sinT + (size_t)pos * HDV;
  float o1 = x1 * cr[d]      - x2 * sr[d];
  float o2 = x2 * cr[d + 40] + x1 * sr[d + 40];
  p[base + d]      = (_Float16)o1;
  p[base + d + 40] = (_Float16)o2;
}

// ------------------------------ tiled WMMA GEMM -----------------------------
// C[M,N] = A_f16[M,K] @ W^T_f16[N,K] (+bias), 128x128 block tiles, 8 waves
// (4x2, each 32x64), K-step 32.  Both tiles are staged with async global->LDS
// b128 copies into double-buffered LDS (layout [row][32] = exactly the WMMA
// fragment layout), pipelined on ASYNCcnt: each lane issues 4 async ops per
// K-step, so waiting asynccnt<=4 retires the current buffer while the next
// buffer's DMA is still in flight.
// MODE 0: scatter into padded Q/K ([seg][head][L][96]) and transposed V
//         ([seg][head][96][L]).  MODE 1: fp32 output to Cout.
template <int MODE>
__global__ __launch_bounds__(256) void k_gemm(
    const _Float16* __restrict__ A, const _Float16* __restrict__ Wt,
    const float* __restrict__ bias, int M, int N, int Kdim,
    _Float16* __restrict__ Qb, _Float16* __restrict__ Kb,
    _Float16* __restrict__ Vt, float* __restrict__ Cout) {
  __shared__ _Float16 At[2][128 * 32];   // A tile  [m:128][k:32]
  __shared__ _Float16 Bt[2][128 * 32];   // W^T tile [n:128][k:32]
  const int tid = threadIdx.x;
  const int w = tid >> 5, lane = tid & 31;
  const int wm = w & 3, wn = w >> 2;          // 4x2 wave grid, wave = 32x64
  const int m0 = blockIdx.y * 128;
  const int n0 = blockIdx.x * 128;
  const int lh  = (lane < 16) ? 0 : 1;
  const int l15 = lane & 15;

  const char* gA = (const char*)(A  + (size_t)m0 * Kdim);
  const char* gB = (const char*)(Wt + (size_t)n0 * Kdim);
  const int r0 = tid >> 2, c0 = (tid & 3) * 8;   // chunk 0: rows 0..63
  const int r1 = r0 + 64;                        // chunk 1: rows 64..127

  v8f acc[2][4];
#pragma unroll
  for (int i = 0; i < 2; i++)
#pragma unroll
    for (int j = 0; j < 4; j++) acc[i][j] = (v8f){0, 0, 0, 0, 0, 0, 0, 0};

  // issue async copies of K-step kt into buffer `buf` (4 b128 ops per lane)
  auto issue = [&](int kt, int buf) {
    int kb = kt * 64;                            // byte offset of K-step
    async_ld16(&At[buf][r0 * 32 + c0], gA, r0 * Kdim * 2 + kb + c0 * 2);
    async_ld16(&At[buf][r1 * 32 + c0], gA, r1 * Kdim * 2 + kb + c0 * 2);
    async_ld16(&Bt[buf][r0 * 32 + c0], gB, r0 * Kdim * 2 + kb + c0 * 2);
    async_ld16(&Bt[buf][r1 * 32 + c0], gB, r1 * Kdim * 2 + kb + c0 * 2);
  };

  const int nk = Kdim / 32;
  issue(0, 0);
  for (int kt = 0; kt < nk; ++kt) {
    const int cur = kt & 1;
    if (kt + 1 < nk) { issue(kt + 1, cur ^ 1); wait_async4(); }
    else             { wait_async0(); }
    __syncthreads();   // all waves' DMA for `cur` visible

    Frag16 a[2], b[4];
#pragma unroll
    for (int mt = 0; mt < 2; mt++) {
      int r = wm * 32 + mt * 16 + l15;
      a[mt].q[0] = *(const uint4*)&At[cur][r * 32 + (lh ? 8 : 0)];
      a[mt].q[1] = *(const uint4*)&At[cur][r * 32 + (lh ? 24 : 16)];
    }
#pragma unroll
    for (int nt = 0; nt < 4; nt++) {
      int n = wn * 64 + nt * 16 + l15;
      b[nt].q[0] = *(const uint4*)&Bt[cur][n * 32 + (lh ? 8 : 0)];
      b[nt].q[1] = *(const uint4*)&Bt[cur][n * 32 + (lh ? 24 : 16)];
    }
#pragma unroll
    for (int mt = 0; mt < 2; mt++)
#pragma unroll
      for (int nt = 0; nt < 4; nt++)
        acc[mt][nt] = __builtin_amdgcn_wmma_f32_16x16x32_f16(
            false, a[mt].h, false, b[nt].h, (short)0, acc[mt][nt], false, false);

    __syncthreads();   // reads of `cur` done before it is refilled
  }

  // epilogue
#pragma unroll
  for (int mt = 0; mt < 2; mt++)
#pragma unroll
    for (int nt = 0; nt < 4; nt++) {
      int n_g = n0 + wn * 64 + nt * 16 + l15;
      float bv = bias[n_g];
#pragma unroll
      for (int i = 0; i < 8; i++) {
        int m_g = m0 + wm * 32 + mt * 16 + i + lh * 8;
        float v = acc[mt][nt][i] + bv;
        if (MODE == 0) {
          int which = n_g / DIMV;            // 0=q 1=k 2=v
          int d = n_g - which * DIMV;
          int head = d / HDV;
          int hd = d - head * HDV;
          int seg = m_g >> 10, r = m_g & 1023;
          size_t sh = (size_t)(seg * HEADSV + head);
          _Float16 hv = (_Float16)v;
          if (which == 0)      Qb[(sh * LSEGV + r) * HDP + hd] = hv;
          else if (which == 1) Kb[(sh * LSEGV + r) * HDP + hd] = hv;
          else                 Vt[(sh * HDP + hd) * LSEGV + r] = hv;
        } else {
          Cout[(size_t)m_g * N + n_g] = v;
        }
      }
    }
}

// ---------------------- attention: one WG per (seg,head,qblk) ---------------
// LDS: S strip 64x1024 f32 (262144B, later overlaid by f16 probs),
//      Q tile 64x96 f16 (12288B), inv-sums (256B). Total 274688B < 320KB.
__global__ __launch_bounds__(256) void k_attn(
    const _Float16* __restrict__ Q, const _Float16* __restrict__ Kb,
    const _Float16* __restrict__ Vt, _Float16* __restrict__ AO) {
  extern __shared__ char smem[];
  float*    S   = (float*)smem;
  _Float16* Qt  = (_Float16*)(smem + 262144);
  float*    inv = (float*)(smem + 274432);

  const int tid = threadIdx.x, w = tid >> 5, lane = tid & 31;
  const int l15 = lane & 15, lh = (lane < 16) ? 0 : 1;
  const int blk = blockIdx.x;             // seg*256 + head*16 + qb
  const int qb = blk & 15, head = (blk >> 4) & 15, seg = blk >> 8;
  const size_t sh = (size_t)(seg * HEADSV + head);
  const _Float16* Qseg = Q  + sh * LSEGV * HDP;
  const _Float16* Kseg = Kb + sh * LSEGV * HDP;
  const _Float16* Vseg = Vt + sh * HDP * LSEGV;
  const int q0 = qb * 64;

  // load Q tile 64x96 into LDS
  for (int j = tid; j < 64 * 12; j += 256) {
    int r = j / 12, c = j % 12;
    *(uint4*)&Qt[r * 96 + c * 8] =
        *(const uint4*)&Qseg[(size_t)(q0 + r) * HDP + c * 8];
  }
  __syncthreads();

  // ---- phase 1: S = scale * Q @ K^T  (4 M-tiles x 64 N-tiles over 8 waves)
  const int mt = w & 3;
  Frag16 a[3];
#pragma unroll
  for (int ks = 0; ks < 3; ks++) {
    int r = mt * 16 + l15, k0 = ks * 32;
    a[ks].q[0] = *(const uint4*)&Qt[r * 96 + k0 + (lh ? 8 : 0)];
    a[ks].q[1] = *(const uint4*)&Qt[r * 96 + k0 + (lh ? 24 : 16)];
  }
  const float scale = 0.11180339887498949f;   // 1/sqrt(80)
  for (int nt = (w >> 2); nt < 64; nt += 2) {
    if (nt + 2 < 64)   // hint next K rows toward the caches
      __builtin_prefetch(&Kseg[(size_t)((nt + 2) * 16 + l15) * HDP], 0, 3);
    v8f sv = (v8f){0, 0, 0, 0, 0, 0, 0, 0};
#pragma unroll
    for (int ks = 0; ks < 3; ks++) {
      Frag16 b;
      int krow = nt * 16 + l15;
      int koff = ks * 32 + (lh ? 8 : 0);
      b.q[0] = *(const uint4*)&Kseg[(size_t)krow * HDP + koff];
      b.q[1] = *(const uint4*)&Kseg[(size_t)krow * HDP + koff + 16];
      sv = __builtin_amdgcn_wmma_f32_16x16x32_f16(false, a[ks].h, false, b.h,
                                                  (short)0, sv, false, false);
    }
#pragma unroll
    for (int i = 0; i < 8; i++) {
      int m = mt * 16 + i + lh * 8;
      int n = nt * 16 + l15;
      S[m * 1024 + n] = sv[i] * scale;
    }
  }
  __syncthreads();

  // ---- phase 2: softmax rows [8w, 8w+8); probs written f16 in place over S.
  // P row r lives at byte 32768*(r>>3) + 2048*(r&7): stays inside wave w's own
  // S rows and never clobbers data still to be read (write-after-read safe).
#pragma unroll 1
  for (int ii = 0; ii < 8; ii++) {
    int r = w * 8 + ii;
    float x[32];
    float mx = -1e30f;
#pragma unroll
    for (int t = 0; t < 32; t++) {
      x[t] = S[r * 1024 + t * 32 + lane];
      mx = fmaxf(mx, x[t]);
    }
    mx = wave_reduce_max(mx);
    asm volatile("" ::: "memory");   // keep f16 stores after all f32 loads
    _Float16* Pr = (_Float16*)(smem + 32768 * (r >> 3) + 2048 * (r & 7));
    float sum = 0.f;
#pragma unroll
    for (int t = 0; t < 32; t++) {
      float e = __expf(x[t] - mx);
      sum += e;
      Pr[t * 32 + lane] = (_Float16)e;
    }
    sum = wave_reduce_sum(sum);
    if (lane == 0) inv[r] = 1.0f / sum;
  }
  __syncthreads();

  // ---- phase 3: O = P @ V  (4 M-tiles x 5 N-tiles; waves = 4 mt x {3,2} nt)
  const int mt3 = w >> 1, nh = w & 1;
  const int ntb = nh * 3, nte = nh ? 5 : 3;
  v8f oacc[3];
#pragma unroll
  for (int i = 0; i < 3; i++) oacc[i] = (v8f){0, 0, 0, 0, 0, 0, 0, 0};
  for (int kk = 0; kk < 32; kk++) {
    int k0 = kk * 32;
    Frag16 a3;
    {
      int r = mt3 * 16 + l15;
      const char* pb = smem + 32768 * (r >> 3) + 2048 * (r & 7);
      a3.q[0] = *(const uint4*)(pb + 2 * (k0 + (lh ? 8 : 0)));
      a3.q[1] = *(const uint4*)(pb + 2 * (k0 + (lh ? 24 : 16)));
    }
    for (int nt = ntb; nt < nte; nt++) {
      Frag16 b;
      int n = nt * 16 + l15;
      int koff = k0 + (lh ? 8 : 0);
      b.q[0] = *(const uint4*)&Vseg[(size_t)n * LSEGV + koff];
      b.q[1] = *(const uint4*)&Vseg[(size_t)n * LSEGV + koff + 16];
      oacc[nt - ntb] = __builtin_amdgcn_wmma_f32_16x16x32_f16(
          false, a3.h, false, b.h, (short)0, oacc[nt - ntb], false, false);
    }
  }
  for (int nt = ntb; nt < nte; nt++) {
#pragma unroll
    for (int i = 0; i < 8; i++) {
      int m = mt3 * 16 + i + lh * 8;
      int n = nt * 16 + l15;
      float v = oacc[nt - ntb][i] * inv[m];
      AO[(size_t)(seg * LSEGV + q0 + m) * DIMV + head * HDV + n] = (_Float16)v;
    }
  }
}

// ------------------------------- host launcher ------------------------------
extern "C" void kernel_launch(void* const* d_in, const int* in_sizes, int n_in,
                              void* d_out, int out_size, void* d_ws,
                              size_t ws_size, hipStream_t stream) {
  const float* hidden = (const float*)d_in[0];
  const float* cosT   = (const float*)d_in[1];
  const float* sinT   = (const float*)d_in[2];
  const float* qkv_w  = (const float*)d_in[3];
  const float* qkv_b  = (const float*)d_in[4];
  const float* proj_w = (const float*)d_in[5];
  const float* proj_b = (const float*)d_in[6];
  float* out = (float*)d_out;
  (void)in_sizes; (void)n_in; (void)out_size; (void)ws_size;

  char* ws = (char*)d_ws;
  const size_t bXh    = (size_t)SEQV * DIMV * 2;                   // 20.97MB
  const size_t bWqkv  = (size_t)DIMV * NQKV * 2;                   //  9.83MB
  const size_t bWproj = (size_t)DIMV * DIMV * 2;                   //  3.28MB
  const size_t bQK    = (size_t)NSEGV * HEADSV * LSEGV * HDP * 2;  // 25.17MB
  _Float16* Xh    = (_Float16*)(ws);
  _Float16* WqkvT = (_Float16*)(ws + bXh);                         // [3840][1280]
  _Float16* WprojT= (_Float16*)(ws + bXh + bWqkv);                 // [1280][1280]
  _Float16* Qb    = (_Float16*)(ws + bXh + bWqkv + bWproj);
  _Float16* Kb    = (_Float16*)(ws + bXh + bWqkv + bWproj + bQK);
  _Float16* Vt    = (_Float16*)(ws + bXh + bWqkv + bWproj + 2 * bQK);
  _Float16* AO    = (_Float16*)(ws + bXh + bWqkv + bWproj + 3 * bQK); // ~130MB total

  // 1) convert activations; weights converted + transposed to [N][K]
  k_f32_to_f16<<<(SEQV * DIMV + 255) / 256, 256, 0, stream>>>(hidden, Xh, SEQV * DIMV);
  k_f32_to_f16_T<<<(DIMV * NQKV + 255) / 256, 256, 0, stream>>>(qkv_w, WqkvT, DIMV, NQKV);
  k_f32_to_f16_T<<<(DIMV * DIMV + 255) / 256, 256, 0, stream>>>(proj_w, WprojT, DIMV, DIMV);

  // 2) zero the padded head-dim columns of Q/K
  k_pad_qk<<<(2 * 131072 * 16) / 256, 256, 0, stream>>>(Qb, Kb);

  // 3) fused QKV GEMM -> padded Q/K + transposed V
  dim3 g1(NQKV / 128, SEQV / 128);
  k_gemm<0><<<g1, 256, 0, stream>>>(Xh, WqkvT, qkv_b, SEQV, NQKV, DIMV,
                                    Qb, Kb, Vt, nullptr);

  // 4) RoPE on Q and K
  k_rope<<<(2 * NSEGV * HEADSV * LSEGV * 40) / 256, 256, 0, stream>>>(Qb, Kb, cosT, sinT);

  // 5) attention (274688B dynamic LDS; CDNA5 WGP has 320KB)
  k_attn<<<NSEGV * HEADSV * 16, 256, 274688, stream>>>(Qb, Kb, Vt, AO);

  // 6) output projection -> fp32 d_out
  dim3 g2(DIMV / 128, SEQV / 128);
  k_gemm<1><<<g2, 256, 0, stream>>>(AO, WprojT, proj_b, SEQV, DIMV, DIMV,
                                    nullptr, nullptr, nullptr, out);
}
